// WorldModel_35278861369588
// MI455X (gfx1250) — matrix operand order repaired and checked by reading, not compiled, not run on hardware
//
#include <hip/hip_runtime.h>
#include <cstdint>

// ---------------------------------------------------------------------------
// CDNA5 (gfx1250) wave32 implementation of the 2-layer retention model.
// GEMMs/attention on v_wmma_f32_16x16x32_bf16 (fp32 accum); global->LDS tile
// movement uses gfx1250 async-to-LDS ops (ASYNCcnt) with double buffering.
// ---------------------------------------------------------------------------

typedef __attribute__((ext_vector_type(16))) __bf16 v16bf;
typedef __attribute__((ext_vector_type(8)))  float  v8f;

union Frag16 { v16bf v; uint32_t u[8]; };
union FragF  { v8f v;   float    f[8]; };

#define BATCH  4
#define SEQ    2048
#define DMODEL 1024
#define NHEAD  8
#define DHEAD  128
#define MTOT   (BATCH * SEQ)          /* 8192 rows */
#define DFFN   (4 * DMODEL)           /* 4096 */

__device__ __forceinline__ uint16_t f32_to_bf16(float f) {
  uint32_t x = __float_as_uint(f);
  uint32_t r = x + 0x7FFFu + ((x >> 16) & 1u);   // round-to-nearest-even
  return (uint16_t)(r >> 16);
}

// Generic pointers to __shared__ carry the LDS byte offset in the low 32 bits
// (ISA: LDS aperture truncates to addr[31:0]).
__device__ __forceinline__ uint32_t lds_addr32(const void* p) {
  return (uint32_t)(uintptr_t)p;
}

// gfx1250 async global->LDS copy, 16 bytes per lane, tracked by ASYNCcnt.
__device__ __forceinline__ void async_b128(uint32_t lds_off, const void* gaddr) {
  asm volatile("global_load_async_to_lds_b128 %0, %1, off"
               :: "v"(lds_off), "v"(gaddr) : "memory");
}
__device__ __forceinline__ void wait_async_le4() {
  asm volatile("s_wait_asynccnt 0x4" ::: "memory");
}
__device__ __forceinline__ void wait_async_0() {
  asm volatile("s_wait_asynccnt 0x0" ::: "memory");
}

// A-fragment (16x32 bf16): lanes 0-15 row M=lane, K = {0..7, 16..23};
// lanes 16-31 row M=lane-16, K = {8..15, 24..31}. p = row base + (lane&16? 8:0)
__device__ __forceinline__ void load_a_frag(Frag16& a, const uint16_t* p) {
  a.u[0] = *(const uint32_t*)(p + 0);
  a.u[1] = *(const uint32_t*)(p + 2);
  a.u[2] = *(const uint32_t*)(p + 4);
  a.u[3] = *(const uint32_t*)(p + 6);
  a.u[4] = *(const uint32_t*)(p + 16);
  a.u[5] = *(const uint32_t*)(p + 18);
  a.u[6] = *(const uint32_t*)(p + 20);
  a.u[7] = *(const uint32_t*)(p + 22);
}

// B-fragment (32x16 bf16): lane holds col n=lane&15; lanes 0-15 K={0..7,16..23},
// lanes 16-31 K={8..15,24..31}. LDS layout [n][k] (k contiguous).
__device__ __forceinline__ void load_b_frag(Frag16& b, const uint16_t* p) {
#pragma unroll
  for (int v = 0; v < 8; v++) b.u[v] = *(const uint32_t*)(p + 2 * v + (v >= 4 ? 8 : 0));
}

__device__ __forceinline__ v8f wmma_bf16(const Frag16& a, const Frag16& b, v8f c) {
  return __builtin_amdgcn_wmma_f32_16x16x32_bf16(false, a.v, false, b.v,
                                                 (short)0, c, false, false);
}

// ---------------------------------------------------------------------------
// f32 -> bf16 straight convert (activations)
// ---------------------------------------------------------------------------
__global__ __launch_bounds__(256) void cvt_f32_bf16(const float* __restrict__ in,
                                                    uint16_t* __restrict__ out,
                                                    int n) {
  int i = blockIdx.x * blockDim.x + threadIdx.x;
  int stride = gridDim.x * blockDim.x;
  for (; i < n; i += stride) out[i] = f32_to_bf16(in[i]);
}

// ---------------------------------------------------------------------------
// f32 [K,N] -> bf16 [N,K] tiled transpose (weights)
// ---------------------------------------------------------------------------
__global__ __launch_bounds__(256) void transpose_cvt(const float* __restrict__ in,
                                                     uint16_t* __restrict__ out,
                                                     int K, int N) {
  __shared__ float t[32][33];
  const int c  = threadIdx.x & 31;
  const int r0 = threadIdx.x >> 5;
  const int n0 = blockIdx.x * 32;
  const int k0 = blockIdx.y * 32;
#pragma unroll
  for (int i = 0; i < 4; i++) {
    int r = r0 + i * 8;
    t[r][c] = in[(size_t)(k0 + r) * N + (n0 + c)];
  }
  __syncthreads();
#pragma unroll
  for (int i = 0; i < 4; i++) {
    int r = r0 + i * 8;
    out[(size_t)(n0 + r) * K + (k0 + c)] = f32_to_bf16(t[c][r]);
  }
}

// ---------------------------------------------------------------------------
// GEMM: C[M,N] = A[M,K](bf16) * Wt[N,K](bf16 pre-transposed) + bias[N].
// 128x128 tile, BK=32, 8 waves x (32x64). Double-buffered LDS fed by async
// global->LDS copies: tile k+1 streams in while WMMAs consume tile k.
// Epilogues: 0 f32 | 1 bf16 | 2 bf16 transposed [b,h,d,t] (V) |
//            3 sigmoid f32 | 4 gelu(erf) bf16 | 5 resid add -> f32 + bf16
// ---------------------------------------------------------------------------
template <int EPI>
__global__ __launch_bounds__(256) void gemm_bf16(
    const uint16_t* __restrict__ A, const uint16_t* __restrict__ Bt,
    const float* __restrict__ bias, const float* resid,
    float* Cf, uint16_t* Cb, int M, int N, int K) {
  __shared__ uint16_t As[2][128 * 40];   // [m][k] stride 40 (pad)
  __shared__ uint16_t Bs[2][128 * 40];   // [n][k] stride 40 (pad)

  const int tid  = threadIdx.x;
  const int lane = tid & 31;
  const int wid  = tid >> 5;
  const int wm   = wid & 3;              // 4 row-blocks of 32
  const int wn   = wid >> 2;             // 2 col-blocks of 64
  const int m0   = blockIdx.y * 128;
  const int n0   = blockIdx.x * 128;

  FragF acc[2][4];
#pragma unroll
  for (int i = 0; i < 2; i++)
#pragma unroll
    for (int j = 0; j < 4; j++)
#pragma unroll
      for (int r = 0; r < 8; r++) acc[i][j].f[r] = 0.0f;

  const int lrow = tid >> 1;             // 0..127
  const int lseg = (tid & 1) * 16;       // 0 / 16
  const uint16_t* Ag = A  + (size_t)(m0 + lrow) * K + lseg;
  const uint16_t* Bg = Bt + (size_t)(n0 + lrow) * K + lseg;
  const uint32_t AsW[2] = {lds_addr32(&As[0][lrow * 40 + lseg]),
                           lds_addr32(&As[1][lrow * 40 + lseg])};
  const uint32_t BsW[2] = {lds_addr32(&Bs[0][lrow * 40 + lseg]),
                           lds_addr32(&Bs[1][lrow * 40 + lseg])};

  const int arow  = wm * 32 + (lane & 15);
  const int akoff = (lane & 16) ? 8 : 0;
  const int bkoff = (lane & 16) ? 8 : 0;

  auto issue = [&](int k0, int buf) {
    async_b128(AsW[buf],      Ag + k0);
    async_b128(AsW[buf] + 16, Ag + k0 + 8);
    async_b128(BsW[buf],      Bg + k0);
    async_b128(BsW[buf] + 16, Bg + k0 + 8);
  };

  const int NK = K >> 5;
  issue(0, 0);
  for (int kt = 0; kt < NK; kt++) {
    if (kt + 1 < NK) {
      issue((kt + 1) * 32, (kt + 1) & 1);
      wait_async_le4();                  // in-order: drains current tile only
    } else {
      wait_async_0();
    }
    __syncthreads();                     // tile kt visible to all waves

    const uint16_t* Ab = As[kt & 1];
    const uint16_t* Bb = Bs[kt & 1];
    Frag16 a[2], b[4];
#pragma unroll
    for (int i = 0; i < 2; i++)
      load_a_frag(a[i], Ab + (arow + i * 16) * 40 + akoff);
#pragma unroll
    for (int f = 0; f < 4; f++)
      load_b_frag(b[f], Bb + (wn * 64 + f * 16 + (lane & 15)) * 40 + bkoff);
#pragma unroll
    for (int i = 0; i < 2; i++)
#pragma unroll
      for (int f = 0; f < 4; f++)
        acc[i][f].v = wmma_bf16(a[i], b[f], acc[i][f].v);
    __syncthreads();                     // all reads of buf done before reuse
  }

  // epilogue: C frag layout: VGPR r -> M = r + (lane&16?8:0), N = lane&15
#pragma unroll
  for (int i = 0; i < 2; i++)
#pragma unroll
    for (int f = 0; f < 4; f++)
#pragma unroll
      for (int r = 0; r < 8; r++) {
        const int m = m0 + wm * 32 + i * 16 + r + ((lane & 16) ? 8 : 0);
        const int n = n0 + wn * 64 + f * 16 + (lane & 15);
        float v = acc[i][f].f[r] + bias[n];
        const size_t idx = (size_t)m * N + n;
        if (EPI == 0) {
          Cf[idx] = v;
        } else if (EPI == 1) {
          Cb[idx] = f32_to_bf16(v);
        } else if (EPI == 2) {  // V: store transposed as [b][h][d][t]
          const int bb = m >> 11, t = m & (SEQ - 1);
          const int hh = n >> 7,  d = n & (DHEAD - 1);
          Cb[(((size_t)(bb * NHEAD + hh) * DHEAD + d) << 11) + t] = f32_to_bf16(v);
        } else if (EPI == 3) {  // sigmoid gate
          Cf[idx] = 1.0f / (1.0f + __expf(-v));
        } else if (EPI == 4) {  // exact-erf GELU
          float g = 0.5f * v * (1.0f + erff(v * 0.70710678118654752f));
          Cb[idx] = f32_to_bf16(g);
        } else {                // residual add; dual write f32 + bf16
          float o = resid[idx] + v;
          Cf[idx] = o;
          Cb[idx] = f32_to_bf16(o);
        }
      }
}

// ---------------------------------------------------------------------------
// Flash-style decayed linear attention (no softmax -> exact streaming).
// grid = (SEQ/64, NHEAD, BATCH).
// decay factored: 0.99^(s-t) = 0.99^(q0-t0) * 0.99^m * 0.99^(-n); only one
// __expf per key tile, causal mask only on the diagonal tile.
// ---------------------------------------------------------------------------
__global__ __launch_bounds__(256) void attention_kernel(
    const uint16_t* __restrict__ Qb, const uint16_t* __restrict__ Kb,
    const uint16_t* __restrict__ Vt, float* __restrict__ ret) {
  __shared__ uint16_t Qs[64 * 136];   // [s][d]
  __shared__ uint16_t Ks[64 * 136];   // [t][d]
  __shared__ uint16_t Vs[128 * 72];   // [d][t]
  __shared__ uint16_t Ss[64 * 72];    // [s][t] bf16 scores*decay

  const int tid  = threadIdx.x;
  const int lane = tid & 31;
  const int wid  = tid >> 5;
  const int qblk = blockIdx.x;
  const int h    = blockIdx.y;
  const int b    = blockIdx.z;
  const int q0   = qblk * 64;
  const int rm   = wid & 3;           // 16-row block
  const int cn   = wid >> 2;          // stage1: 32-col blk; stage2: 64-col blk
  const int hi8  = (lane & 16) ? 8 : 0;
  const int nloc = lane & 15;

  const float lnD   = -0.010050335853501441f;  // ln(0.99)
  const float scale = 0.08838834764831845f;    // 1/sqrt(128)
  static const float pw[8] = {1.0f, 0.99f, 0.9801f, 0.970299f, 0.96059601f,
                              0.9509900499f, 0.941480149401f, 0.93206534790699f};
  // per-wave decay factors (hoisted out of the key loop)
  const float Em_base = __expf((float)(rm * 16 + hi8) * lnD);     // 0.99^m_base
  const float En0 = __expf(-(float)(cn * 32 + nloc) * lnD);       // 0.99^-n (f=0)
  const float En1 = En0 * __expf(16.0f * 0.010050335853501441f);  // f=1

  {  // async-load Q tile 64x128
    const int row = tid >> 2, seg = (tid & 3) * 32;
    const uint16_t* g = Qb + (size_t)(b * SEQ + q0 + row) * DMODEL + h * DHEAD + seg;
    const uint32_t s = lds_addr32(&Qs[row * 136 + seg]);
#pragma unroll
    for (int j = 0; j < 4; j++) async_b128(s + j * 16, g + j * 8);
  }

  FragF rc[4];
#pragma unroll
  for (int f = 0; f < 4; f++)
#pragma unroll
    for (int r = 0; r < 8; r++) rc[f].f[r] = 0.0f;

  const int akoff = (lane & 16) ? 8 : 0;
  const int bkoff = (lane & 16) ? 8 : 0;

  for (int kt = 0; kt <= qblk; kt++) {
    const int t0 = kt * 64;
    {  // async K tile 64x128
      const int row = tid >> 2, seg = (tid & 3) * 32;
      const uint16_t* g = Kb + (size_t)(b * SEQ + t0 + row) * DMODEL + h * DHEAD + seg;
      const uint32_t s = lds_addr32(&Ks[row * 136 + seg]);
#pragma unroll
      for (int j = 0; j < 4; j++) async_b128(s + j * 16, g + j * 8);
    }
    {  // async V tile 128x64 from transposed global [b,h,d,t]
      const int d = tid >> 1, seg = (tid & 1) * 32;
      const uint16_t* g = Vt + (((size_t)(b * NHEAD + h) * DHEAD + d) << 11) + t0 + seg;
      const uint32_t s = lds_addr32(&Vs[d * 72 + seg]);
#pragma unroll
      for (int j = 0; j < 4; j++) async_b128(s + j * 16, g + j * 8);
    }
    wait_async_0();
    __syncthreads();

    // ---- stage 1: S = Q * K^T (wave: rows rm*16, cols cn*32 .. +31) ----
    FragF sacc[2];
#pragma unroll
    for (int f = 0; f < 2; f++)
#pragma unroll
      for (int r = 0; r < 8; r++) sacc[f].f[r] = 0.0f;
#pragma unroll
    for (int ks = 0; ks < 4; ks++) {  // d = ks*32
      Frag16 a;
      load_a_frag(a, Qs + (rm * 16 + (lane & 15)) * 136 + ks * 32 + akoff);
#pragma unroll
      for (int f = 0; f < 2; f++) {
        Frag16 bfr;
        load_b_frag(bfr, Ks + (cn * 32 + f * 16 + (lane & 15)) * 136 + ks * 32 + bkoff);
        sacc[f].v = wmma_bf16(a, bfr, sacc[f].v);
      }
    }
    // decay epilogue -> bf16 scores in LDS (1 exp per tile, 1 mul per elem)
    {
      const float Cf_ = scale * __expf((float)(q0 - t0) * lnD);
      const float wb0 = Cf_ * Em_base * En0;
      const float wb1 = Cf_ * Em_base * En1;
      if (kt < qblk) {  // strictly-lower tile: no mask needed
#pragma unroll
        for (int f = 0; f < 2; f++)
#pragma unroll
          for (int r = 0; r < 8; r++) {
            const int m = rm * 16 + hi8 + r;
            const int n = cn * 32 + f * 16 + nloc;
            const float w = (f ? wb1 : wb0) * pw[r];
            Ss[m * 72 + n] = f32_to_bf16(sacc[f].f[r] * w);
          }
      } else {          // diagonal tile: causal mask m >= n
#pragma unroll
        for (int f = 0; f < 2; f++)
#pragma unroll
          for (int r = 0; r < 8; r++) {
            const int m = rm * 16 + hi8 + r;
            const int n = cn * 32 + f * 16 + nloc;
            float w = (f ? wb1 : wb0) * pw[r];
            if (m < n) w = 0.0f;
            Ss[m * 72 + n] = f32_to_bf16(sacc[f].f[r] * w);
          }
      }
    }
    __syncthreads();

    // ---- stage 2: ret += S * V (wave: rows rm*16, cols cn*64 .. +63) ----
#pragma unroll
    for (int ks = 0; ks < 2; ks++) {  // t = ks*32
      Frag16 a;
      load_a_frag(a, Ss + (rm * 16 + (lane & 15)) * 72 + ks * 32 + akoff);
#pragma unroll
      for (int f = 0; f < 4; f++) {
        Frag16 bfr;
        load_b_frag(bfr, Vs + (cn * 64 + f * 16 + (lane & 15)) * 72 + ks * 32 + bkoff);
        rc[f].v = wmma_bf16(a, bfr, rc[f].v);
      }
    }
    __syncthreads();
  }

  // store ret tile (f32, row-major [m, h*128+n])
#pragma unroll
  for (int f = 0; f < 4; f++)
#pragma unroll
    for (int r = 0; r < 8; r++) {
      const int m = rm * 16 + hi8 + r;
      const int n = cn * 64 + f * 16 + nloc;
      ret[(size_t)(b * SEQ + q0 + m) * DMODEL + h * DHEAD + n] = rc[f].f[r];
    }
}

// ---------------------------------------------------------------------------
// LayerNorm over D=1024. One row per block, 256 threads x 4 elems.
// MODE 0: out bf16 | MODE 1: out bf16 * G(gate,f32) | MODE 2: out f32
// ---------------------------------------------------------------------------
template <int MODE>
__global__ __launch_bounds__(256) void layernorm_kernel(
    const float* __restrict__ x, const float* __restrict__ g,
    const float* __restrict__ bta, const float* __restrict__ gmul,
    uint16_t* __restrict__ ob, float* __restrict__ of) {
  __shared__ float red[256];
  const int row = blockIdx.x, tid = threadIdx.x;
  const float* xr = x + (size_t)row * DMODEL;
  float v[4];
#pragma unroll
  for (int i = 0; i < 4; i++) v[i] = xr[tid + i * 256];

  red[tid] = v[0] + v[1] + v[2] + v[3];
  __syncthreads();
  for (int off = 128; off > 0; off >>= 1) {
    if (tid < off) red[tid] += red[tid + off];
    __syncthreads();
  }
  const float mean = red[0] * (1.0f / DMODEL);
  __syncthreads();

  float q = 0.0f;
#pragma unroll
  for (int i = 0; i < 4; i++) { float d = v[i] - mean; q += d * d; }
  red[tid] = q;
  __syncthreads();
  for (int off = 128; off > 0; off >>= 1) {
    if (tid < off) red[tid] += red[tid + off];
    __syncthreads();
  }
  const float rstd = rsqrtf(red[0] * (1.0f / DMODEL) + 1e-5f);

#pragma unroll
  for (int i = 0; i < 4; i++) {
    const int idx = tid + i * 256;
    const float o = (v[i] - mean) * rstd * g[idx] + bta[idx];
    const size_t gi = (size_t)row * DMODEL + idx;
    if (MODE == 0)      ob[gi] = f32_to_bf16(o);
    else if (MODE == 1) ob[gi] = f32_to_bf16(o * gmul[gi]);
    else                of[gi] = o;
  }
}

// ---------------------------------------------------------------------------
// Host orchestration
// ---------------------------------------------------------------------------
extern "C" void kernel_launch(void* const* d_in, const int* in_sizes, int n_in,
                              void* d_out, int out_size, void* d_ws, size_t ws_size,
                              hipStream_t stream) {
  const float* x     = (const float*)d_in[0];
  const float* win_w = (const float*)d_in[1];
  const float* win_b = (const float*)d_in[2];
  const float* ln_g  = (const float*)d_in[3];
  const float* ln_b  = (const float*)d_in[4];
  const float* wq_w  = (const float*)d_in[5];
  const float* wq_b  = (const float*)d_in[6];
  const float* wk_w  = (const float*)d_in[7];
  const float* wk_b  = (const float*)d_in[8];
  const float* wv_w  = (const float*)d_in[9];
  const float* wv_b  = (const float*)d_in[10];
  const float* wg_w  = (const float*)d_in[11];
  const float* wg_b  = (const float*)d_in[12];
  const float* wo_w  = (const float*)d_in[13];
  const float* wo_b  = (const float*)d_in[14];
  const float* rn_g  = (const float*)d_in[15];
  const float* rn_b  = (const float*)d_in[16];
  const float* f1_w  = (const float*)d_in[17];
  const float* f1_b  = (const float*)d_in[18];
  const float* f2_w  = (const float*)d_in[19];
  const float* f2_b  = (const float*)d_in[20];
  const float* on_g  = (const float*)d_in[21];
  const float* on_b  = (const float*)d_in[22];

  char* base = (char*)d_ws;
  size_t off = 0;
  auto alloc = [&](size_t bytes) -> void* {
    off = (off + 255) & ~(size_t)255;
    void* p = base + off;
    off += bytes;
    return p;
  };

  const size_t EL = (size_t)MTOT * DMODEL;  // 8M
  uint16_t* xb   = (uint16_t*)alloc(EL * 2);
  uint16_t* wtin = (uint16_t*)alloc((size_t)DMODEL * DMODEL * 2);
  uint16_t *wtq[2], *wtk[2], *wtv[2], *wtg[2], *wto[2], *wtf1[2], *wtf2[2];
  for (int l = 0; l < 2; l++) {
    wtq[l]  = (uint16_t*)alloc((size_t)DMODEL * DMODEL * 2);
    wtk[l]  = (uint16_t*)alloc((size_t)DMODEL * DMODEL * 2);
    wtv[l]  = (uint16_t*)alloc((size_t)DMODEL * DMODEL * 2);
    wtg[l]  = (uint16_t*)alloc((size_t)DMODEL * DMODEL * 2);
    wto[l]  = (uint16_t*)alloc((size_t)DMODEL * DMODEL * 2);
    wtf1[l] = (uint16_t*)alloc((size_t)DMODEL * DFFN * 2);
    wtf2[l] = (uint16_t*)alloc((size_t)DFFN * DMODEL * 2);
  }
  float*    hbuf = (float*)alloc(EL * 4);
  uint16_t* hb   = (uint16_t*)alloc(EL * 2);
  uint16_t* yb   = (uint16_t*)alloc(EL * 2);
  uint16_t* Qbuf = (uint16_t*)alloc(EL * 2);
  uint16_t* Kbuf = (uint16_t*)alloc(EL * 2);
  uint16_t* Vtb  = (uint16_t*)alloc(EL * 2);
  float*    Gf   = (float*)alloc(EL * 4);
  float*    retf = (float*)alloc(EL * 4);
  uint16_t* ogb  = (uint16_t*)alloc(EL * 2);
  // ub (8192x4096 bf16 = 64MB) aliases the dead Q/K/V/G region (80MB)
  uint16_t* ub = Qbuf;

  // ---- weight + activation conversion ----
  cvt_f32_bf16<<<2048, 256, 0, stream>>>(x, xb, (int)EL);
  {
    dim3 tg(DMODEL / 32, DMODEL / 32);
    transpose_cvt<<<tg, 256, 0, stream>>>(win_w, wtin, DMODEL, DMODEL);
    for (int l = 0; l < 2; l++) {
      const size_t so = (size_t)l * DMODEL * DMODEL;
      transpose_cvt<<<tg, 256, 0, stream>>>(wq_w + so, wtq[l], DMODEL, DMODEL);
      transpose_cvt<<<tg, 256, 0, stream>>>(wk_w + so, wtk[l], DMODEL, DMODEL);
      transpose_cvt<<<tg, 256, 0, stream>>>(wv_w + so, wtv[l], DMODEL, DMODEL);
      transpose_cvt<<<tg, 256, 0, stream>>>(wg_w + so, wtg[l], DMODEL, DMODEL);
      transpose_cvt<<<tg, 256, 0, stream>>>(wo_w + so, wto[l], DMODEL, DMODEL);
      transpose_cvt<<<dim3(DFFN / 32, DMODEL / 32), 256, 0, stream>>>(
          f1_w + (size_t)l * DMODEL * DFFN, wtf1[l], DMODEL, DFFN);
      transpose_cvt<<<dim3(DMODEL / 32, DFFN / 32), 256, 0, stream>>>(
          f2_w + (size_t)l * DFFN * DMODEL, wtf2[l], DFFN, DMODEL);
    }
  }

  const dim3 gD(DMODEL / 128, MTOT / 128);   // (8, 64)
  const dim3 gF(DFFN / 128, MTOT / 128);     // (32, 64)

  // h = x @ win + b
  gemm_bf16<0><<<gD, 256, 0, stream>>>(xb, wtin, win_b, nullptr, hbuf, nullptr,
                                       MTOT, DMODEL, DMODEL);

  for (int l = 0; l < 2; l++) {
    const float* lg = ln_g + l * DMODEL;
    const float* lb = ln_b + l * DMODEL;
    layernorm_kernel<0><<<MTOT, 256, 0, stream>>>(hbuf, lg, lb, nullptr, yb, nullptr);

    gemm_bf16<1><<<gD, 256, 0, stream>>>(yb, wtq[l], wq_b + l * DMODEL, nullptr,
                                         nullptr, Qbuf, MTOT, DMODEL, DMODEL);
    gemm_bf16<1><<<gD, 256, 0, stream>>>(yb, wtk[l], wk_b + l * DMODEL, nullptr,
                                         nullptr, Kbuf, MTOT, DMODEL, DMODEL);
    gemm_bf16<2><<<gD, 256, 0, stream>>>(yb, wtv[l], wv_b + l * DMODEL, nullptr,
                                         nullptr, Vtb, MTOT, DMODEL, DMODEL);
    gemm_bf16<3><<<gD, 256, 0, stream>>>(yb, wtg[l], wg_b + l * DMODEL, nullptr,
                                         Gf, nullptr, MTOT, DMODEL, DMODEL);

    attention_kernel<<<dim3(SEQ / 64, NHEAD, BATCH), 256, 0, stream>>>(
        Qbuf, Kbuf, Vtb, retf);

    // og = LN(ret) * sigmoid_gate  (bf16)
    layernorm_kernel<1><<<MTOT, 256, 0, stream>>>(retf, rn_g + l * DMODEL,
                                                  rn_b + l * DMODEL, Gf, ogb, nullptr);
    // h = h + og @ wo + b   (also refresh bf16 copy hb)
    gemm_bf16<5><<<gD, 256, 0, stream>>>(ogb, wto[l], wo_b + l * DMODEL, hbuf,
                                         hbuf, hb, MTOT, DMODEL, DMODEL);
    // u = gelu(h @ f1 + b)
    gemm_bf16<4><<<gF, 256, 0, stream>>>(hb, wtf1[l], f1_b + l * DFFN, nullptr,
                                         nullptr, ub, MTOT, DFFN, DMODEL);
    // h = h + u @ f2 + b
    gemm_bf16<5><<<gD, 256, 0, stream>>>(ub, wtf2[l], f2_b + l * DMODEL, hbuf,
                                         hbuf, hb, MTOT, DMODEL, DFFN);
  }

  layernorm_kernel<2><<<MTOT, 256, 0, stream>>>(hbuf, on_g, on_b, nullptr, nullptr,
                                                (float*)d_out);
}
